// DecoderBlock_52587579572824
// MI455X (gfx1250) — matrix operand order repaired and checked by reading, not compiled, not run on hardware
//
#include <hip/hip_runtime.h>
#include <math.h>

// Problem constants (from reference)
#define B_  16
#define T_  512
#define C_  768
#define H_  12
#define HS_ 64
#define F_  3840          // 5*C
#define EPS_ 1e-5f

typedef __attribute__((ext_vector_type(16))) __bf16   bf16x16;
typedef __attribute__((ext_vector_type(2)))  __bf16   bf16x2;
typedef __attribute__((ext_vector_type(8)))  float    f32x8;
typedef __attribute__((ext_vector_type(4)))  unsigned u32x4;
typedef __attribute__((ext_vector_type(8)))  unsigned u32x8;

// fp32 -> bf16 via V_FMA_MIXLO_BF16 (CDNA5 VOP3P op 62): cvt(fma(x,1,0)), RNE
__device__ __forceinline__ __bf16 f2bf(float f) {
  unsigned d = 0;
  asm("v_fma_mixlo_bf16 %0, %1, %2, %3"
      : "+v"(d) : "v"(f), "v"(1.0f), "v"(0.0f));
  return __builtin_bit_cast(bf16x2, d)[0];
}
// packed pair: mixlo into [15:0], mixhi into [31:16]
__device__ __forceinline__ bf16x2 f2bf2(float a, float b) {
  unsigned d = 0;
  asm("v_fma_mixlo_bf16 %0, %1, %3, %4\n\t"
      "v_fma_mixhi_bf16 %0, %2, %3, %4"
      : "+v"(d) : "v"(a), "v"(b), "v"(1.0f), "v"(0.0f));
  return __builtin_bit_cast(bf16x2, d);
}

// batch offset map: offset(z) = (z/div)*outer + (z%div)*inner   (element units)
struct BMap { long outer; long inner; int div; };
__device__ __forceinline__ long boff(BMap m, int z) {
  return (long)(z / m.div) * m.outer + (long)(z % m.div) * m.inner;
}

// ---------------------------------------------------------------------------
// Tensor Data Mover: async 2D tile (tileX elems of 2B) x (tileY rows) from
// global (row stride = strideElems) into contiguous LDS at ldsOff.
// D# built per CDNA5 ISA (2D: VADDR2/3 = NULL).
// ---------------------------------------------------------------------------
__device__ __forceinline__ void tdm_load_2d(unsigned ldsOff, const void* gptr,
                                            unsigned strideElems,
                                            unsigned tileX, unsigned tileY) {
  unsigned long long ga = (unsigned long long)(uintptr_t)gptr;
  u32x4 g0;
  g0[0] = 1u;                                        // count=1, user mode
  g0[1] = ldsOff;                                    // lds_addr (bytes)
  g0[2] = (unsigned)ga;                              // global_addr[31:0]
  g0[3] = (unsigned)((ga >> 32) & 0x01FFFFFFu)       // global_addr[56:32]
        | (2u << 30);                                // type = 2 ("image")
  u32x8 g1;
  g1[0] = 1u << 16;                                  // data_size = 1 (2 bytes)
  g1[1] = (tileX & 0xFFFFu) << 16;                   // tensor_dim0[15:0]
  g1[2] = (tileX >> 16) | ((tileY & 0xFFFFu) << 16); // dim0 hi | tensor_dim1 lo
  g1[3] = (tileY >> 16) | ((tileX & 0xFFFFu) << 16); // dim1 hi | tile_dim0
  g1[4] = tileY & 0xFFFFu;                           // tile_dim1 (tile_dim2 = 0)
  g1[5] = strideElems;                               // tensor_dim0_stride lo32
  g1[6] = 0;                                         // stride hi | dim1_stride lo
  g1[7] = 0;                                         // dim1_stride hi (2D: unused)
  asm volatile("tensor_load_to_lds %0, %1" :: "s"(g0), "s"(g1) : "memory");
}

// ---------------------------------------------------------------------------
// Generic batched bf16 GEMM: out = scale*(A(MxK)*Bt^T + bias) [gelu] [+resid]
// A row-major bf16 (lda); B given transposed: Bt (N x K row-major, ldb).
// Wave tile = (16*TM) x (16*TN) of v_wmma_f32_16x16x32_bf16; block = WM x WN
// waves. Both tiles staged global->LDS by TDM, double-buffered: wave 0 issues
// next K-step's descriptors while the block computes on the current buffer.
// ---------------------------------------------------------------------------
template <int WM, int WN, int TM, int TN, bool DO_GELU>
__global__ void __launch_bounds__(WM * WN * 32)
gemm_bf16_kernel(const __bf16* __restrict__ A, BMap mA, int lda,
                 const __bf16* __restrict__ Bt, BMap mB, int ldb,
                 const float* __restrict__ bias,
                 const float* __restrict__ resid, BMap mR, int ldr,
                 float* __restrict__ outF, __bf16* __restrict__ outB,
                 BMap mO, int ldo, float outScale,
                 int M, int N, int K) {
  constexpr int BM = WM * 16 * TM, BN = WN * 16 * TN, KS = 32;
  __shared__ __bf16 lA[2][BM][KS];
  __shared__ __bf16 lB[2][BN][KS];

  const int z = blockIdx.z;
  const __bf16* Ab = A + boff(mA, z);
  const __bf16* Bb = Bt + boff(mB, z);
  const int tid  = threadIdx.x;
  const int wave = tid >> 5, lane = tid & 31;
  const int half = lane >> 4, l16 = lane & 15;
  const int wm = wave % WM, wn = wave / WM;
  const int tM = blockIdx.y * BM, tN = blockIdx.x * BN;
  const unsigned ldsA[2] = {(unsigned)(uintptr_t)&lA[0][0][0],
                            (unsigned)(uintptr_t)&lA[1][0][0]};
  const unsigned ldsB[2] = {(unsigned)(uintptr_t)&lB[0][0][0],
                            (unsigned)(uintptr_t)&lB[1][0][0]};

  f32x8 acc[TM][TN] = {};

  // prologue: stage first tiles
  if (wave == 0) {
    tdm_load_2d(ldsA[0], Ab + (long)tM * lda, (unsigned)lda, KS, BM);
    tdm_load_2d(ldsB[0], Bb + (long)tN * ldb, (unsigned)ldb, KS, BN);
    __builtin_amdgcn_s_wait_tensorcnt(0);
  }
  __syncthreads();

  int cur = 0;
  for (int k0 = 0; k0 < K; k0 += KS) {
    const int nk = k0 + KS;
    if (nk < K && wave == 0) {     // async prefetch next K-step into alt buffer
      tdm_load_2d(ldsA[cur ^ 1], Ab + (long)tM * lda + nk, (unsigned)lda, KS, BM);
      tdm_load_2d(ldsB[cur ^ 1], Bb + (long)tN * ldb + nk, (unsigned)ldb, KS, BN);
    }

    // fragments: A 16x32 / B 32x16, both in contiguous k order (k = 16*half + j)
    bf16x16 af[TM], bfg[TN];
    #pragma unroll
    for (int i = 0; i < TM; ++i) {
      const __bf16* ar = &lA[cur][wm * 16 * TM + i * 16 + l16][0];
      #pragma unroll
      for (int j = 0; j < 16; ++j) {
        int p = j >> 1;
        int kk = j + 16 * half;
        af[i][j] = ar[kk];
      }
    }
    #pragma unroll
    for (int i = 0; i < TN; ++i) {
      const __bf16* br = &lB[cur][wn * 16 * TN + i * 16 + l16][16 * half];
      #pragma unroll
      for (int j = 0; j < 16; ++j) bfg[i][j] = br[j];
    }
    #pragma unroll
    for (int i = 0; i < TM; ++i)
      #pragma unroll
      for (int j = 0; j < TN; ++j)
        acc[i][j] = __builtin_amdgcn_wmma_f32_16x16x32_bf16(
            false, af[i], false, bfg[j], (short)0, acc[i][j], false, false);

    if (wave == 0) __builtin_amdgcn_s_wait_tensorcnt(0);
    __syncthreads();
    cur ^= 1;
  }

  // epilogue: bias/scale/gelu/residual, fp32 and/or bf16 store
  const float* Rb = resid ? resid + boff(mR, z) : nullptr;
  const long Ob = boff(mO, z);
  #pragma unroll
  for (int i = 0; i < TM; ++i) {
    #pragma unroll
    for (int j = 0; j < TN; ++j) {
      #pragma unroll
      for (int r = 0; r < 8; ++r) {
        int m = tM + wm * 16 * TM + i * 16 + r + 8 * half;
        int n = tN + wn * 16 * TN + j * 16 + l16;
        float v = acc[i][j][r];
        if (bias) v += bias[n];
        v *= outScale;
        if (DO_GELU) v = 0.5f * v * (1.0f + erff(v * 0.70710678118654752f));
        if (Rb) v += Rb[(long)m * ldr + n];
        long o = Ob + (long)m * ldo + n;
        if (outF) outF[o] = v;
        if (outB) outB[o] = f2bf(v);
      }
    }
  }
}

// ---------------------------------------------------------------------------
// Relative-position bias: for fixed (h,t): scores[b, h, t, s] += Q[b,:]·R[s,:]
// 16 x 512 x 64 WMMA GEMM over all 16 batches; rel_pos fp32 streamed once,
// converted in-register with v_fma_mixlo/mixhi_bf16.
// ---------------------------------------------------------------------------
__global__ void __launch_bounds__(128)
relbias_kernel(const __bf16* __restrict__ q, const float* __restrict__ rel,
               float* __restrict__ scores) {
  const int t = blockIdx.x, h = blockIdx.y;
  __shared__ __bf16 lq[16][64];
  const int tid = threadIdx.x;
  {
    int b = tid >> 3, cc = (tid & 7) * 8;
    *(uint4*)&lq[b][cc] =
        *(const uint4*)(q + ((long)b * T_ + t) * C_ + h * HS_ + cc);
  }
  __syncthreads();
  const int wave = tid >> 5, lane = tid & 31, half = lane >> 4, l16 = lane & 15;
  const float* rb = rel + (((long)h * T_ + t) * T_) * HS_;
  f32x8 acc[8] = {};
  #pragma unroll
  for (int k0 = 0; k0 < HS_; k0 += 32) {
    bf16x16 af;
    const __bf16* ar = &lq[l16][k0];
    #pragma unroll
    for (int j = 0; j < 16; ++j) {
      int p = j >> 1;
      int kk = j + 16 * half;
      af[j] = ar[kk];
    }
    #pragma unroll
    for (int nt = 0; nt < 8; ++nt) {
      int s = wave * 128 + nt * 16 + l16;
      const float* rr = rb + (long)s * HS_ + k0 + 16 * half;
      union { bf16x16 v; bf16x2 p[8]; } bu;
      #pragma unroll
      for (int j = 0; j < 8; ++j) bu.p[j] = f2bf2(rr[2 * j], rr[2 * j + 1]);
      acc[nt] = __builtin_amdgcn_wmma_f32_16x16x32_bf16(
          false, af, false, bu.v, (short)0, acc[nt], false, false);
    }
  }
  #pragma unroll
  for (int nt = 0; nt < 8; ++nt) {
    int s = wave * 128 + nt * 16 + l16;
    #pragma unroll
    for (int r = 0; r < 8; ++r) {
      int b = r + 8 * half;
      float* p = scores + (((long)b * H_ + h) * T_ + t) * T_ + s;
      *p += acc[nt][r];
    }
  }
}

// ---------------------------------------------------------------------------
// RMSNorm row kernel: fp32 in -> bf16 out
// ---------------------------------------------------------------------------
__global__ void __launch_bounds__(256)
rmsnorm_kernel(const float* __restrict__ x, const float* __restrict__ w,
               __bf16* __restrict__ out) {
  const long row = blockIdx.x;
  const float* xr = x + row * C_;
  const int tid = threadIdx.x;
  __shared__ float red[256];
  float s = 0.f;
  for (int i = tid; i < C_; i += 256) { float v = xr[i]; s += v * v; }
  red[tid] = s; __syncthreads();
  for (int k = 128; k > 0; k >>= 1) {
    if (tid < k) red[tid] += red[tid + k];
    __syncthreads();
  }
  float scale = rsqrtf(red[0] / (float)C_ + EPS_);
  for (int i = tid; i < C_; i += 256) out[row * C_ + i] = f2bf(xr[i] * scale * w[i]);
}

// ---------------------------------------------------------------------------
// Softmax over one score row (len T=512); rewrites fp32 row in place as
// packed bf16 attention probabilities. masked=1 => causal.
// ---------------------------------------------------------------------------
__global__ void __launch_bounds__(256)
softmax_kernel(float* __restrict__ scores, int masked) {
  const long row = blockIdx.x;
  const int t = (int)(row % T_);
  float* sr = scores + row * (long)T_;
  const int tid = threadIdx.x;
  __shared__ float red[256];
  float v0 = sr[2 * tid], v1 = sr[2 * tid + 1];
  if (masked) {
    if (2 * tid > t) v0 = -INFINITY;
    if (2 * tid + 1 > t) v1 = -INFINITY;
  }
  float mx = fmaxf(v0, v1);
  red[tid] = mx; __syncthreads();
  for (int k = 128; k > 0; k >>= 1) {
    if (tid < k) red[tid] = fmaxf(red[tid], red[tid + k]);
    __syncthreads();
  }
  mx = red[0]; __syncthreads();
  float e0 = __expf(v0 - mx), e1 = __expf(v1 - mx);
  red[tid] = e0 + e1; __syncthreads();
  for (int k = 128; k > 0; k >>= 1) {
    if (tid < k) red[tid] += red[tid + k];
    __syncthreads();
  }
  float inv = 1.0f / red[0];
  __syncthreads();          // all fp32 reads done before aliased bf16 writes
  ((bf16x2*)sr)[tid] = f2bf2(e0 * inv, e1 * inv);
}

// ---------------------------------------------------------------------------
// V transpose: (b,t,h,d) bf16 -> (b,h,d,t) bf16 so att@V's B operand is NxK.
// 12.6 MB, L2-resident; output-linear indexing.
// ---------------------------------------------------------------------------
__global__ void vtrans_kernel(const __bf16* __restrict__ v,
                              __bf16* __restrict__ vT) {
  long i = (long)blockIdx.x * 256 + threadIdx.x;
  if (i >= (long)B_ * C_ * T_) return;
  int t = (int)(i % T_); long r = i / T_;   // r = (b*H + h)*HS + d
  int d = (int)(r % HS_); long bh = r / HS_;
  int h = (int)(bh % H_); long b = bh / H_;
  vT[i] = v[(b * T_ + t) * C_ + h * HS_ + d];
}

// ---------------------------------------------------------------------------
// Weight repack kernels: fp32 -> bf16, transposed to (N x K) row-major
// ---------------------------------------------------------------------------
__global__ void wt_hcd_kernel(const float* __restrict__ W, __bf16* __restrict__ Wt) {
  long i = (long)blockIdx.x * 256 + threadIdx.x;     // i = n*C + k
  if (i >= (long)C_ * C_) return;
  int n = (int)(i / C_), k = (int)(i % C_);
  int h = n >> 6, d = n & 63;
  Wt[i] = f2bf(W[((long)h * C_ + k) * HS_ + d]);
}
__global__ void wt_kn_kernel(const float* __restrict__ W, __bf16* __restrict__ Wt,
                             int K, int N) {
  long i = (long)blockIdx.x * 256 + threadIdx.x;     // i = n*K + k
  if (i >= (long)K * N) return;
  int n = (int)(i / K), k = (int)(i % K);
  Wt[i] = f2bf(W[(long)k * N + n]);
}

// ---------------------------------------------------------------------------
extern "C" void kernel_launch(void* const* d_in, const int* in_sizes, int n_in,
                              void* d_out, int out_size, void* d_ws, size_t ws_size,
                              hipStream_t stream) {
  (void)in_sizes; (void)n_in; (void)out_size; (void)ws_size;
  const float* x     = (const float*)d_in[0];
  const float* nw    = (const float*)d_in[1];
  const float* Wq    = (const float*)d_in[2];
  const float* bq    = (const float*)d_in[3];
  const float* Wk    = (const float*)d_in[4];
  const float* bk    = (const float*)d_in[5];
  const float* Wv    = (const float*)d_in[6];
  const float* bv    = (const float*)d_in[7];
  const float* rel   = (const float*)d_in[8];
  const float* Wproj = (const float*)d_in[9];
  const float* bproj = (const float*)d_in[10];
  const float* W1    = (const float*)d_in[11];
  const float* b1    = (const float*)d_in[12];
  const float* W2    = (const float*)d_in[13];
  const float* b2    = (const float*)d_in[14];
  float* out = (float*)d_out;

  // workspace carve (256B aligned)
  char* p = (char*)d_ws;
  auto carve = [&](size_t bytes) -> void* {
    void* r = (void*)p; p += (bytes + 255) & ~(size_t)255; return r;
  };
  const size_t MT = (size_t)B_ * T_;
  __bf16* WqT = (__bf16*)carve((size_t)C_ * C_ * 2);
  __bf16* WkT = (__bf16*)carve((size_t)C_ * C_ * 2);
  __bf16* WvT = (__bf16*)carve((size_t)C_ * C_ * 2);
  __bf16* WpT = (__bf16*)carve((size_t)C_ * C_ * 2);
  __bf16* W1T = (__bf16*)carve((size_t)C_ * F_ * 2);
  __bf16* W2T = (__bf16*)carve((size_t)C_ * F_ * 2);
  __bf16* xnb = (__bf16*)carve(MT * C_ * 2);
  __bf16* qb  = (__bf16*)carve(MT * C_ * 2);
  __bf16* kb  = (__bf16*)carve(MT * C_ * 2);
  __bf16* vb  = (__bf16*)carve(MT * C_ * 2);
  __bf16* vTb = (__bf16*)carve(MT * C_ * 2);
  __bf16* cat = (__bf16*)carve(MT * C_ * 2);
  __bf16* hb  = (__bf16*)carve(MT * F_ * 2);
  float*  x1  = (float*)carve(MT * C_ * 4);
  float*  x2  = (float*)carve(MT * C_ * 4);
  float*  scores = (float*)carve((size_t)B_ * H_ * T_ * T_ * 4);

  const BMap triv{0, 0, 1};

  // repack weights (bf16, N x K) — deterministic per launch
  wt_hcd_kernel<<<(C_ * C_ + 255) / 256, 256, 0, stream>>>(Wq, WqT);
  wt_hcd_kernel<<<(C_ * C_ + 255) / 256, 256, 0, stream>>>(Wk, WkT);
  wt_hcd_kernel<<<(C_ * C_ + 255) / 256, 256, 0, stream>>>(Wv, WvT);
  wt_kn_kernel<<<(C_ * C_ + 255) / 256, 256, 0, stream>>>(Wproj, WpT, C_, C_);
  wt_kn_kernel<<<(C_ * F_ + 255) / 256, 256, 0, stream>>>(W1, W1T, C_, F_);
  wt_kn_kernel<<<(C_ * F_ + 255) / 256, 256, 0, stream>>>(W2, W2T, F_, C_);

  auto attn = [&](const float* xin, float* xout, bool masked) {
    rmsnorm_kernel<<<B_ * T_, 256, 0, stream>>>(xin, nw, xnb);
    // QKV projections: (BT x C) * (C x C); sqrt(HS)=8 folded into K
    dim3 g1(C_ / 128, (int)MT / 64, 1);
    gemm_bf16_kernel<2, 2, 2, 4, false><<<g1, 128, 0, stream>>>(
        xnb, triv, C_, WqT, triv, C_, bq, nullptr, triv, 0,
        nullptr, qb, triv, C_, 1.0f, (int)MT, C_, C_);
    gemm_bf16_kernel<2, 2, 2, 4, false><<<g1, 128, 0, stream>>>(
        xnb, triv, C_, WkT, triv, C_, bk, nullptr, triv, 0,
        nullptr, kb, triv, C_, 8.0f, (int)MT, C_, C_);
    gemm_bf16_kernel<2, 2, 2, 4, false><<<g1, 128, 0, stream>>>(
        xnb, triv, C_, WvT, triv, C_, bv, nullptr, triv, 0,
        nullptr, vb, triv, C_, 1.0f, (int)MT, C_, C_);
    vtrans_kernel<<<(int)((MT * C_ + 255) / 256), 256, 0, stream>>>(vb, vTb);
    // scores[b,h] = Q(TxHS) * K(TxHS)^T, batched over z = b*H + h
    const BMap mqk{(long)T_ * C_, (long)HS_, H_};
    const BMap msc{(long)T_ * T_, 0, 1};
    dim3 g2(T_ / 128, T_ / 64, B_ * H_);
    gemm_bf16_kernel<2, 2, 2, 4, false><<<g2, 128, 0, stream>>>(
        qb, mqk, C_, kb, mqk, C_, nullptr, nullptr, triv, 0,
        scores, nullptr, msc, T_, 1.0f, T_, T_, HS_);
    if (!masked)
      relbias_kernel<<<dim3(T_, H_), 128, 0, stream>>>(qb, rel, scores);
    softmax_kernel<<<B_ * H_ * T_, 256, 0, stream>>>(scores, masked ? 1 : 0);
    // out[b,h] = att(TxT) * V(TxHS); att bf16 in-place (row pitch 2T),
    // V transposed so B operand is NxK (TDM path)
    const BMap matt{(long)T_ * 2 * T_, 0, 1};
    const BMap mvT{(long)C_ * T_, (long)HS_ * T_, H_};
    const BMap mo{(long)T_ * C_, (long)HS_, H_};
    dim3 g3(HS_ / 32, T_ / 128, B_ * H_);
    gemm_bf16_kernel<2, 2, 4, 1, false><<<g3, 128, 0, stream>>>(
        (const __bf16*)(const void*)scores, matt, 2 * T_, vTb, mvT, T_,
        nullptr, nullptr, triv, 0, nullptr, cat, mo, C_, 1.0f, T_, HS_, T_);
    // output projection + residual -> fp32 xout
    gemm_bf16_kernel<2, 2, 2, 4, false><<<g1, 128, 0, stream>>>(
        cat, triv, C_, WpT, triv, C_, bproj, xin, triv, C_,
        xout, nullptr, triv, C_, 1.0f, (int)MT, C_, C_);
  };

  attn(x, x1, true);    // pass 1: causal
  attn(x1, x2, false);  // pass 2: rel-pos bias

  // FFN
  rmsnorm_kernel<<<B_ * T_, 256, 0, stream>>>(x2, nw, xnb);
  gemm_bf16_kernel<2, 2, 2, 4, true><<<dim3(F_ / 128, (int)MT / 64, 1), 128, 0, stream>>>(
      xnb, triv, C_, W1T, triv, C_, b1, nullptr, triv, 0,
      nullptr, hb, triv, F_, 1.0f, (int)MT, F_, C_);
  gemm_bf16_kernel<2, 2, 2, 4, false><<<dim3(C_ / 128, (int)MT / 64, 1), 128, 0, stream>>>(
      hb, triv, F_, W2T, triv, F_, b2, x2, triv, C_,
      out, nullptr, triv, C_, 1.0f, (int)MT, C_, F_);
}
